// DCNSepHead_55637006352908
// MI455X (gfx1250) — matrix operand order repaired and checked by reading, not compiled, not run on hardware
//
#include <hip/hip_runtime.h>

typedef __attribute__((ext_vector_type(16))) _Float16 v16h;
typedef __attribute__((ext_vector_type(8)))  _Float16 v8h;
typedef __attribute__((ext_vector_type(8)))  float    v8f;

#define BB 4
#define HH 128
#define WW 128

// LDS byte offset of a generic pointer that points into __shared__ memory.
__device__ __forceinline__ unsigned lds_addr_of(const void* p) {
    return (unsigned)(unsigned long long)(const __attribute__((address_space(3))) void*)p;
}

// Async copy of 16 bytes global -> LDS (tracked by ASYNCcnt).
__device__ __forceinline__ void async_copy_b128(unsigned lds_off, const void* gaddr) {
    asm volatile("global_load_async_to_lds_b128 %0, %1, off"
                 :: "v"(lds_off), "v"(gaddr)
                 : "memory");
}

__device__ __forceinline__ void wait_asynccnt0() {
    asm volatile("s_wait_asynccnt 0x0" ::: "memory");
}

// ---- WMMA A-fragment loader: wpk is [9][Mpad][64] f16, row-major in c.
// 16x32 f16 A layout (contiguous, matching B): lanes 0-15 hold K = Kb+{0..15},
// lanes 16-31 hold K = Kb+{16..31}; i.e. two contiguous 8-half chunks.
__device__ __forceinline__ v16h load_a_frag(const _Float16* __restrict__ wpk,
                                            int t, int Mpad, int m, int kb, int laneHi) {
    const _Float16* base = wpk + ((size_t)(t * Mpad + m)) * 64 + kb * 32 + 16 * laneHi;
    v8h lo = *(const v8h*)(base);
    v8h hi = *(const v8h*)(base + 8);
    return __builtin_shufflevector(lo, hi, 0,1,2,3,4,5,6,7,8,9,10,11,12,13,14,15);
}

// ================= pack kernels =================
__global__ __launch_bounds__(256) void nchw_to_nhwc_kernel(
    const float* __restrict__ x, _Float16* __restrict__ xn) {
    int idx = blockIdx.x * 256 + threadIdx.x;            // B*H*W*64 = 4194304
    int c = idx & 63, w = (idx >> 6) & 127, h = (idx >> 13) & 127, b = idx >> 20;
    xn[idx] = (_Float16)x[(((size_t)(b * 64 + c)) * HH + h) * WW + w];
}

__global__ __launch_bounds__(256) void pack3x3_kernel(
    const float* __restrict__ src, _Float16* __restrict__ dst, int Oreal, int Mpad) {
    int idx = blockIdx.x * 256 + threadIdx.x;
    int total = 9 * Mpad * 64;
    if (idx >= total) return;
    int t = idx / (Mpad * 64);
    int rem = idx - t * (Mpad * 64);
    int o = rem >> 6, c = rem & 63;
    float v = (o < Oreal) ? src[((size_t)(o * 64 + c)) * 9 + t] : 0.f;
    dst[idx] = (_Float16)v;
}

// ================= fused offset-conv + deformable-conv (DCNv1) =================
// grid (W/64, H, B), 256 threads = 8 waves. Output: ReLU(dcn) in NHWC f16.
__global__ __launch_bounds__(256) void dcn_fused_kernel(
    const _Float16* __restrict__ xn,   // [B,H,W,64] f16
    const float*    __restrict__ offw, // [72,64]
    const float*    __restrict__ offb, // [72]
    const _Float16* __restrict__ wpk,  // [9,64,64] f16
    _Float16*       __restrict__ feat) // [B,H,W,64] f16
{
    __shared__ alignas(32) float     xv[64][64];      // 16 KB
    __shared__ alignas(32) float     off[64][72];     // 18 KB
    __shared__ alignas(32) _Float16  btile[64][64];   //  8 KB

    const int tid = threadIdx.x;
    const int lane = tid & 31;
    const int wid = tid >> 5;
    const int laneLo = lane & 15, laneHi = lane >> 4;
    const int b = blockIdx.z, h = blockIdx.y, w0 = blockIdx.x * 64;

    // warm L2/L0 with this wave's packed A rows
    __builtin_prefetch(wpk + ((size_t)((wid >> 2) * 16 + laneLo)) * 64, 0, 0);

    // ---- stage x tile (f16 -> f32 LDS), 256 threads cover 64px * 4 chunks
    {
        int p = tid >> 2, q = tid & 3;
        const v16h xc = *(const v16h*)(xn + (((size_t)(b * HH + h)) * WW + w0 + p) * 64 + q * 16);
        #pragma unroll
        for (int i = 0; i < 16; ++i) xv[p][q * 16 + i] = (float)xc[i];
    }
    __syncthreads();

    // ---- inline 1x1 offset conv: 64px * 72 outputs
    for (int idx = tid; idx < 64 * 72; idx += 256) {
        int p = idx / 72, j = idx - p * 72;
        float acc = offb[j];
        const float* wr = offw + j * 64;
        #pragma unroll 8
        for (int c = 0; c < 64; ++c) acc += wr[c] * xv[p][c];
        off[p][j] = acc;
    }
    __syncthreads();

    v8f acc[2] = {};
    for (int t = 0; t < 9; ++t) {
        const int di = t / 3 - 1, dj = t % 3 - 1;
        // ---- bilinear sample B-tile [64 K(chan)][64 N(px)], stored [p][c]
        {
            int p = tid >> 2, dg = tid & 3;
            float dy = off[p][(dg * 9 + t) * 2 + 0];
            float dx = off[p][(dg * 9 + t) * 2 + 1];
            float py = (float)(h + di) + dy;
            float px = (float)(w0 + p + dj) + dx;
            float y0f = floorf(py), x0f = floorf(px);
            float fy = py - y0f, fx = px - x0f;
            int y0 = (int)y0f, x0 = (int)x0f;
            float s[16];
            #pragma unroll
            for (int i = 0; i < 16; ++i) s[i] = 0.f;
            #pragma unroll
            for (int corner = 0; corner < 4; ++corner) {
                int yy = y0 + (corner >> 1);
                int xx = x0 + (corner & 1);
                float wgt = ((corner >> 1) ? fy : 1.f - fy) * ((corner & 1) ? fx : 1.f - fx);
                if (yy >= 0 && yy < HH && xx >= 0 && xx < WW) {
                    const v16h cv = *(const v16h*)(xn + (((size_t)(b * HH + yy)) * WW + xx) * 64 + dg * 16);
                    #pragma unroll
                    for (int i = 0; i < 16; ++i) s[i] += wgt * (float)cv[i];
                }
            }
            v16h o16;
            #pragma unroll
            for (int i = 0; i < 16; ++i) o16[i] = (_Float16)s[i];
            *(v16h*)&btile[p][dg * 16] = o16;
        }
        __syncthreads();
        // ---- GEMM accumulate: each wave owns 2 of 16 (Mtile,Ntile) tiles
        #pragma unroll
        for (int s2 = 0; s2 < 2; ++s2) {
            int ti = wid + s2 * 8;
            int tm = ti >> 2, tn = ti & 3;
            int m = tm * 16 + laneLo;
            #pragma unroll
            for (int kb = 0; kb < 2; ++kb) {
                v16h a = load_a_frag(wpk, t, 64, m, kb, laneHi);
                v16h bf = *(const v16h*)&btile[tn * 16 + laneLo][kb * 32 + 16 * laneHi];
                acc[s2] = __builtin_amdgcn_wmma_f32_16x16x32_f16(
                    false, a, false, bf, (short)0, acc[s2], false, false);
            }
        }
        __syncthreads();
    }

    // ---- ReLU epilogue -> NHWC f16 (8 contiguous channels per lane)
    #pragma unroll
    for (int s2 = 0; s2 < 2; ++s2) {
        int ti = wid + s2 * 8;
        int tm = ti >> 2, tn = ti & 3;
        int pn = tn * 16 + laneLo;
        int ch0 = tm * 16 + 8 * laneHi;
        v8h o;
        #pragma unroll
        for (int r = 0; r < 8; ++r) {
            float v = acc[s2][r];
            o[r] = (_Float16)(v > 0.f ? v : 0.f);
        }
        *(v8h*)(feat + (((size_t)(b * HH + h)) * WW + w0 + pn) * 64 + ch0) = o;
    }
}

// ================= 3x3 conv head kernel =================
// mode 0: +b1, BN(eval), ReLU -> NHWC f16 fout          (cls stem)
// mode 1: +b2 -> fp32 NCHW gout (Mpad=16, oc channels)  (heatmap)
// mode 2: +b1, ReLU, then fused 1x1 (w2,b2) -> fp32 NCHW gout (SepHead branch)
__global__ __launch_bounds__(256) void conv3_head_kernel(
    const _Float16* __restrict__ fin,  // [B,H,W,64] f16
    const _Float16* __restrict__ wpk,  // [9,Mpad,64] f16
    int mode, int Mpad,
    const float* __restrict__ b1,
    const float* __restrict__ bng, const float* __restrict__ bnb,
    const float* __restrict__ bnm, const float* __restrict__ bnv,
    _Float16* __restrict__ fout,
    const float* __restrict__ w2, const float* __restrict__ b2,
    float* __restrict__ gout, int oc)
{
    __shared__ alignas(32) _Float16 rowbuf[3][66][64];  // ~25 KB
    __shared__ alignas(32) float    tbuf[64][64];       //  16 KB (mode 2)

    const int tid = threadIdx.x;
    const int lane = tid & 31;
    const int wid = tid >> 5;
    const int laneLo = lane & 15, laneHi = lane >> 4;
    const int b = blockIdx.z, h = blockIdx.y, w0 = blockIdx.x * 64;

    // warm L2/L0 with this wave's packed A rows
    __builtin_prefetch(wpk + ((size_t)((wid >> 2) * 16 + laneLo)) * 64, 0, 0);

    // ---- stage 3 input rows (halo of 1 each side) via async global->LDS DMA;
    //      zero-pad out-of-bounds halo chunks with plain LDS stores.
    for (int idx = tid; idx < 3 * 66 * 4; idx += 256) {
        int pos = idx >> 2, q = idx & 3;
        int di = pos / 66, wi = pos - di * 66;
        int hh2 = h + di - 1, wwp = w0 + wi - 1;
        if (hh2 >= 0 && hh2 < HH && wwp >= 0 && wwp < WW) {
            const _Float16* g = fin + (((size_t)(b * HH + hh2)) * WW + wwp) * 64 + q * 16;
            async_copy_b128(lds_addr_of(&rowbuf[di][wi][q * 16]), g);
        } else {
            v16h z;
            #pragma unroll
            for (int i = 0; i < 16; ++i) z[i] = (_Float16)0.f;
            *(v16h*)&rowbuf[di][wi][q * 16] = z;
        }
    }
    wait_asynccnt0();
    __syncthreads();

    const int nTiles = ((mode == 1) ? 1 : 4) * 4;
    for (int ti = wid; ti < nTiles; ti += 8) {
        int tm = ti >> 2, tn = ti & 3;
        v8f acc = {};
        for (int t = 0; t < 9; ++t) {
            int di = t / 3, dj = t % 3;
            #pragma unroll
            for (int kb = 0; kb < 2; ++kb) {
                v16h a = load_a_frag(wpk, t, Mpad, tm * 16 + laneLo, kb, laneHi);
                v16h bf = *(const v16h*)&rowbuf[di][tn * 16 + laneLo + dj][kb * 32 + 16 * laneHi];
                acc = __builtin_amdgcn_wmma_f32_16x16x32_f16(
                    false, a, false, bf, (short)0, acc, false, false);
            }
        }
        int pn = tn * 16 + laneLo;
        int ch0 = tm * 16 + 8 * laneHi;
        if (mode == 0) {
            v8h o;
            #pragma unroll
            for (int r = 0; r < 8; ++r) {
                int ch = ch0 + r;
                float sc = bng[ch] * rsqrtf(bnv[ch] + 1e-5f);
                float v = (acc[r] + b1[ch] - bnm[ch]) * sc + bnb[ch];
                o[r] = (_Float16)(v > 0.f ? v : 0.f);
            }
            *(v8h*)(fout + (((size_t)(b * HH + h)) * WW + w0 + pn) * 64 + ch0) = o;
        } else if (mode == 1) {
            #pragma unroll
            for (int r = 0; r < 8; ++r) {
                int o = 8 * laneHi + r;
                if (o < oc)
                    gout[(((size_t)(b * oc + o)) * HH + h) * WW + w0 + pn] = acc[r] + b2[o];
            }
        } else {
            #pragma unroll
            for (int r = 0; r < 8; ++r) {
                float v = acc[r] + b1[ch0 + r];
                tbuf[pn][ch0 + r] = v > 0.f ? v : 0.f;
            }
        }
    }
    if (mode == 2) {
        __syncthreads();
        for (int idx = tid; idx < 64 * oc; idx += 256) {
            int p = idx / oc, o = idx - p * oc;
            float sum = b2[o];
            const float* wr = w2 + o * 64;
            #pragma unroll 8
            for (int c = 0; c < 64; ++c) sum += wr[c] * tbuf[p][c];
            gout[(((size_t)(b * oc + o)) * HH + h) * WW + w0 + p] = sum;
        }
    }
}

// ================= launch =================
extern "C" void kernel_launch(void* const* d_in, const int* in_sizes, int n_in,
                              void* d_out, int out_size, void* d_ws, size_t ws_size,
                              hipStream_t stream) {
    (void)in_sizes; (void)n_in; (void)out_size; (void)ws_size;
    const float* x        = (const float*)d_in[0];
    const float* ca_off_w = (const float*)d_in[1];
    const float* ca_off_b = (const float*)d_in[2];
    const float* ca_dcn_w = (const float*)d_in[3];
    const float* ra_off_w = (const float*)d_in[4];
    const float* ra_off_b = (const float*)d_in[5];
    const float* ra_dcn_w = (const float*)d_in[6];
    const float* cls_w1 = (const float*)d_in[7];
    const float* cls_b1 = (const float*)d_in[8];
    const float* bn_g = (const float*)d_in[9];
    const float* bn_b = (const float*)d_in[10];
    const float* bn_m = (const float*)d_in[11];
    const float* bn_v = (const float*)d_in[12];
    const float* cls_w2 = (const float*)d_in[13];
    const float* cls_b2 = (const float*)d_in[14];
    float* outp = (float*)d_out;

    char* ws = (char*)d_ws;
    size_t off = 0;
    auto alloc = [&](size_t bytes) {
        void* p = ws + off;
        off += (bytes + 255) & ~(size_t)255;
        return p;
    };
    const size_t featBytes = (size_t)BB * HH * WW * 64 * sizeof(_Float16); // 8 MB
    _Float16* xn    = (_Float16*)alloc(featBytes);
    _Float16* featC = (_Float16*)alloc(featBytes);
    _Float16* featR = (_Float16*)alloc(featBytes);
    _Float16* clsH  = (_Float16*)alloc(featBytes);
    _Float16* wpk_ca   = (_Float16*)alloc(9 * 64 * 64 * sizeof(_Float16));
    _Float16* wpk_ra   = (_Float16*)alloc(9 * 64 * 64 * sizeof(_Float16));
    _Float16* wpk_cls1 = (_Float16*)alloc(9 * 64 * 64 * sizeof(_Float16));
    _Float16* wpk_cls2 = (_Float16*)alloc(9 * 16 * 64 * sizeof(_Float16));
    _Float16* wpk_br[5];
    for (int i = 0; i < 5; ++i) wpk_br[i] = (_Float16*)alloc(9 * 64 * 64 * sizeof(_Float16));

    // pack x and weights
    nchw_to_nhwc_kernel<<<(BB * HH * WW * 64) / 256, 256, 0, stream>>>(x, xn);
    pack3x3_kernel<<<(9 * 64 * 64 + 255) / 256, 256, 0, stream>>>(ca_dcn_w, wpk_ca, 64, 64);
    pack3x3_kernel<<<(9 * 64 * 64 + 255) / 256, 256, 0, stream>>>(ra_dcn_w, wpk_ra, 64, 64);
    pack3x3_kernel<<<(9 * 64 * 64 + 255) / 256, 256, 0, stream>>>(cls_w1, wpk_cls1, 64, 64);
    pack3x3_kernel<<<(9 * 16 * 64 + 255) / 256, 256, 0, stream>>>(cls_w2, wpk_cls2, 2, 16);
    for (int i = 0; i < 5; ++i) {
        const float* w1 = (const float*)d_in[15 + 4 * i];
        pack3x3_kernel<<<(9 * 64 * 64 + 255) / 256, 256, 0, stream>>>(w1, wpk_br[i], 64, 64);
    }

    dim3 grid(WW / 64, HH, BB);
    // feature adaption (offset conv + DCN + ReLU), both paths
    dcn_fused_kernel<<<grid, 256, 0, stream>>>(xn, ca_off_w, ca_off_b, wpk_ca, featC);
    dcn_fused_kernel<<<grid, 256, 0, stream>>>(xn, ra_off_w, ra_off_b, wpk_ra, featR);

    // cls head: conv3x3 + BN + ReLU, then conv3x3 -> heatmap
    conv3_head_kernel<<<grid, 256, 0, stream>>>(featC, wpk_cls1, 0, 64,
        cls_b1, bn_g, bn_b, bn_m, bn_v, clsH, nullptr, nullptr, nullptr, 64);
    conv3_head_kernel<<<grid, 256, 0, stream>>>(clsH, wpk_cls2, 1, 16,
        nullptr, nullptr, nullptr, nullptr, nullptr, nullptr, nullptr, cls_b2, outp + 0, 2);

    // SepHead branches: conv3x3 + ReLU + fused conv1x1, fp32 NCHW into d_out
    const int    ocs[5]  = {2, 1, 3, 2, 2};
    const size_t offs[5] = {131072, 262144, 327680, 524288, 655360};
    for (int i = 0; i < 5; ++i) {
        const float* b1 = (const float*)d_in[16 + 4 * i];
        const float* w2 = (const float*)d_in[17 + 4 * i];
        const float* b2 = (const float*)d_in[18 + 4 * i];
        conv3_head_kernel<<<grid, 256, 0, stream>>>(featR, wpk_br[i], 2, 64,
            b1, nullptr, nullptr, nullptr, nullptr, nullptr, w2, b2, outp + offs[i], ocs[i]);
    }
}
